// IFANERCRFModel_74912819577404
// MI455X (gfx1250) — matrix operand order repaired and checked by reading, not compiled, not run on hardware
//
#include <hip/hip_runtime.h>
#include <hip/hip_bf16.h>

typedef __attribute__((ext_vector_type(16))) __bf16 v16bf;
typedef __attribute__((ext_vector_type(8)))  float  v8f;

#define BATCH 64
#define SEQ   512
#define HDIM  1024
#define LBL   25
#define MROWS (BATCH * SEQ)      // 32768
#define NEGF  (-1e30f)

// ---------------- workspace layout (bytes) ----------------
#define EM_BYTES   ((size_t)MROWS * LBL * 4)          // 3,276,800
#define HIST_BYTES ((size_t)BATCH * SEQ * LBL * 4)    // 3,276,800

// =====================================================================
// Kernel 1: emissions GEMM  em = hidden @ fc_w^T + fc_b   (WMMA bf16)
// Block: 256 threads (8 waves). Each wave: 2 M-tiles of 16 rows,
// both N-tiles (cols 0-15, 16-24). Grid: 2048 tiles / 16 = 128 blocks.
// =====================================================================
__global__ __launch_bounds__(256) void emis_gemm_kernel(
    const float* __restrict__ hidden, const float* __restrict__ fc_w,
    const float* __restrict__ fc_b, float* __restrict__ em)
{
  // LDS: fc_w as bf16, chunked layout: chunk c = k8*32 + n holds 8 bf16
  // (n, k8*8 .. k8*8+7).  32 * 1024 * 2B = 64KB, conflict-free 16B reads.
  __shared__ __bf16 Bs[32 * 1024];

  const int tid = threadIdx.x;
  for (int c = tid; c < 4096; c += 256) {
    const int n  = c & 31;
    const int k  = (c >> 5) << 3;
    __bf16* dst = &Bs[c * 8];
    if (n < LBL) {
      const float4 v0 = *(const float4*)(fc_w + n * HDIM + k);
      const float4 v1 = *(const float4*)(fc_w + n * HDIM + k + 4);
      dst[0]=(__bf16)v0.x; dst[1]=(__bf16)v0.y; dst[2]=(__bf16)v0.z; dst[3]=(__bf16)v0.w;
      dst[4]=(__bf16)v1.x; dst[5]=(__bf16)v1.y; dst[6]=(__bf16)v1.z; dst[7]=(__bf16)v1.w;
    } else {
      #pragma unroll
      for (int j = 0; j < 8; ++j) dst[j] = (__bf16)0.0f;
    }
  }
  __syncthreads();

  const int wave = tid >> 5;
  const int lane = tid & 31;
  const int half = lane >> 4;   // which half-wave
  const int nloc = lane & 15;   // M index for A, N index for B/C

  for (int r = 0; r < 2; ++r) {
    const int m0 = (blockIdx.x * 16 + wave * 2 + r) * 16;
    const float* __restrict__ arow = hidden + (size_t)(m0 + nloc) * HDIM;

    v8f c0 = {}; v8f c1 = {};
    for (int k = 0; k < HDIM; k += 32) {
      // ---- A fragment: 16x32 bf16. lane half 0: K k+0..7 / k+16..23,
      //                              half 1: K k+8..15 / k+24..31
      const int ka0 = k + half * 8;
      const int ka1 = k + 16 + half * 8;
      const float4 a0 = *(const float4*)(arow + ka0);
      const float4 a1 = *(const float4*)(arow + ka0 + 4);
      const float4 a2 = *(const float4*)(arow + ka1);
      const float4 a3 = *(const float4*)(arow + ka1 + 4);
      v16bf af;
      af[0]=(__bf16)a0.x; af[1]=(__bf16)a0.y; af[2]=(__bf16)a0.z; af[3]=(__bf16)a0.w;
      af[4]=(__bf16)a1.x; af[5]=(__bf16)a1.y; af[6]=(__bf16)a1.z; af[7]=(__bf16)a1.w;
      af[8]=(__bf16)a2.x; af[9]=(__bf16)a2.y; af[10]=(__bf16)a2.z; af[11]=(__bf16)a2.w;
      af[12]=(__bf16)a3.x; af[13]=(__bf16)a3.y; af[14]=(__bf16)a3.z; af[15]=(__bf16)a3.w;

      // ---- B fragments: 32x16 bf16 per N-tile. lanes 0-15: K k..k+7 / k+16..23,
      //                   lanes 16-31: K k+8..15 / k+24..31, column = nloc(+16)
      const int kb = k + half * 8;
      const __bf16* p0 = &Bs[(((kb >> 3)    ) * 32 + nloc) * 8];
      const __bf16* p1 = &Bs[(((kb >> 3) + 2) * 32 + nloc) * 8];
      const __bf16* q0 = &Bs[(((kb >> 3)    ) * 32 + nloc + 16) * 8];
      const __bf16* q1 = &Bs[(((kb >> 3) + 2) * 32 + nloc + 16) * 8];
      v16bf bf0, bf1;
      #pragma unroll
      for (int e = 0; e < 8; ++e) {
        bf0[e] = p0[e]; bf0[8 + e] = p1[e];
        bf1[e] = q0[e]; bf1[8 + e] = q1[e];
      }

      c0 = __builtin_amdgcn_wmma_f32_16x16x32_bf16(false, af, false, bf0,
                                                   (short)0, c0, false, false);
      c1 = __builtin_amdgcn_wmma_f32_16x16x32_bf16(false, af, false, bf1,
                                                   (short)0, c1, false, false);
    }

    // C layout: VGPR v -> row m0 + v + 8*half, col = nloc (+16 for tile1)
    #pragma unroll
    for (int v = 0; v < 8; ++v) {
      const int row = m0 + v + 8 * half;
      em[(size_t)row * LBL + nloc] = c0[v] + fc_b[nloc];
      const int n1 = nloc + 16;
      if (n1 < LBL) em[(size_t)row * LBL + n1] = c1[v] + fc_b[n1];
    }
  }
}

// =====================================================================
// Kernel 2: CRF forward passes. One wave per (role, batch).
// role 0 (blocks   0- 63): viterbi forward, writes hist + last tag
// role 1 (blocks  64-127): normalizer (forward logsumexp), writes den
// role 2 (blocks 128-191): gold path score, writes num
// =====================================================================
__global__ __launch_bounds__(32) void crf_forward_kernel(
    const float* __restrict__ em, const int* __restrict__ mask,
    const int* __restrict__ labels, const float* __restrict__ start,
    const float* __restrict__ trans, const float* __restrict__ endt,
    int* __restrict__ hist, float* __restrict__ den,
    float* __restrict__ num, int* __restrict__ last_tag)
{
  const int lane = threadIdx.x;
  const int role = blockIdx.x >> 6;
  const int b    = blockIdx.x & 63;
  const float* __restrict__ eb = em + (size_t)b * SEQ * LBL;
  const int*   __restrict__ mb = mask + b * SEQ;

  if (role == 2) {
    if (lane == 0) {
      const int* lb = labels + b * SEQ;
      int prev = lb[0];
      float sc = start[prev] + eb[prev];
      for (int t = 1; t < SEQ; ++t) {
        const int tt = lb[t];
        if (mb[t]) { sc += trans[prev * LBL + tt] + eb[t * LBL + tt]; prev = tt; }
      }
      num[b] = sc + endt[prev];
    }
    return;
  }

  // transition column for this lane: tcol[p] = trans[p][lane]
  float tcol[LBL];
  #pragma unroll
  for (int p = 0; p < LBL; ++p) tcol[p] = (lane < LBL) ? trans[p * LBL + lane] : 0.0f;

  float sc = (lane < LBL) ? (start[lane] + eb[lane]) : NEGF;

  if (role == 0) {
    // -------- Viterbi forward --------
    int* __restrict__ hb = hist + (size_t)b * SEQ * LBL;
    for (int t = 1; t < SEQ; ++t) {
      const float e = (lane < LBL) ? eb[t * LBL + lane] : 0.0f;
      const int   m = mb[t];
      float best = NEGF; int bidx = 0;
      #pragma unroll
      for (int p = 0; p < LBL; ++p) {
        const float cand = __shfl(sc, p, 32) + tcol[p];
        if (cand > best) { best = cand; bidx = p; }
      }
      if (lane < LBL) hb[t * LBL + lane] = bidx;
      const float nxt = best + e;
      sc = (lane < LBL) ? (m ? nxt : sc) : NEGF;
    }
    // argmax(sc + end) over lanes (first-max wins)
    float fv = (lane < LBL) ? (sc + endt[lane]) : NEGF;
    int   fi = lane;
    #pragma unroll
    for (int off = 16; off >= 1; off >>= 1) {
      const float ov = __shfl_down(fv, (unsigned)off, 32);
      const int   oi = __shfl_down(fi, (unsigned)off, 32);
      if (ov > fv) { fv = ov; fi = oi; }
    }
    if (lane == 0) last_tag[b] = fi;
  } else {
    // -------- Normalizer (forward logsumexp) --------
    for (int t = 1; t < SEQ; ++t) {
      const float e = (lane < LBL) ? eb[t * LBL + lane] : 0.0f;
      const int   m = mb[t];
      float cand[LBL];
      float mx = NEGF;
      #pragma unroll
      for (int p = 0; p < LBL; ++p) {
        cand[p] = __shfl(sc, p, 32) + tcol[p];
        mx = fmaxf(mx, cand[p]);
      }
      float s = 0.0f;
      #pragma unroll
      for (int p = 0; p < LBL; ++p) s += __expf(cand[p] - mx);
      const float nxt = mx + __logf(s) + e;
      sc = (lane < LBL) ? (m ? nxt : sc) : NEGF;
    }
    // den = logsumexp(sc + end) across lanes
    float v = (lane < LBL) ? (sc + endt[lane]) : NEGF;
    float mx = v;
    #pragma unroll
    for (int off = 16; off >= 1; off >>= 1)
      mx = fmaxf(mx, __shfl_xor(mx, (unsigned)off, 32));
    float s = __expf(v - mx);
    #pragma unroll
    for (int off = 16; off >= 1; off >>= 1)
      s += __shfl_xor(s, (unsigned)off, 32);
    if (lane == 0) den[b] = mx + __logf(s);
  }
}

// =====================================================================
// Kernel 3: viterbi backtrack (one thread per batch) + loss
// =====================================================================
__global__ __launch_bounds__(64) void crf_backtrack_kernel(
    const int* __restrict__ mask, const int* __restrict__ hist,
    const int* __restrict__ last_tag, const float* __restrict__ den,
    const float* __restrict__ num, float* __restrict__ out)
{
  const int b = threadIdx.x;
  if (b == 0) {
    float acc = 0.0f;
    for (int i = 0; i < BATCH; ++i) acc += num[i] - den[i];
    out[(size_t)BATCH * SEQ] = -acc / (float)BATCH;
  }
  const int* __restrict__ hb = hist + (size_t)b * SEQ * LBL;
  const int* __restrict__ mb = mask + b * SEQ;
  int tag = last_tag[b];
  out[(size_t)b * SEQ + SEQ - 1] = (float)tag;
  for (int t = SEQ - 2; t >= 0; --t) {
    if (mb[t + 1]) tag = hb[(t + 1) * LBL + tag];
    out[(size_t)b * SEQ + t] = (float)tag;
  }
}

// =====================================================================
extern "C" void kernel_launch(void* const* d_in, const int* in_sizes, int n_in,
                              void* d_out, int out_size, void* d_ws, size_t ws_size,
                              hipStream_t stream) {
  (void)in_sizes; (void)n_in; (void)out_size; (void)ws_size;
  const float* hidden = (const float*)d_in[0];
  const float* fc_w   = (const float*)d_in[1];
  const float* fc_b   = (const float*)d_in[2];
  const float* start  = (const float*)d_in[3];
  const float* trans  = (const float*)d_in[4];
  const float* endt   = (const float*)d_in[5];
  const int*   mask   = (const int*)d_in[6];
  const int*   labels = (const int*)d_in[7];
  float* out = (float*)d_out;

  char* ws = (char*)d_ws;
  float* em   = (float*)ws;
  int*   hist = (int*)(ws + EM_BYTES);
  float* den  = (float*)(ws + EM_BYTES + HIST_BYTES);
  float* num  = den + BATCH;
  int*   last = (int*)(num + BATCH);

  // 2048 M-tiles / (8 waves * 2 tiles) = 128 blocks
  emis_gemm_kernel<<<128, 256, 0, stream>>>(hidden, fc_w, fc_b, em);
  crf_forward_kernel<<<192, 32, 0, stream>>>(em, mask, labels, start, trans,
                                             endt, hist, den, num, last);
  crf_backtrack_kernel<<<1, 64, 0, stream>>>(mask, hist, last, den, num, out);
}
